// PositionConditionerBlock_22222160789915
// MI455X (gfx1250) — matrix operand order repaired and checked by reading, not compiled, not run on hardware
//
#include <hip/hip_runtime.h>
#include <hip/hip_bf16.h>

typedef __attribute__((ext_vector_type(16))) _Float16 v16h;
typedef __attribute__((ext_vector_type(8)))  _Float16 v8h;
typedef __attribute__((ext_vector_type(8)))  float    v8f;
typedef __attribute__((ext_vector_type(4)))  float    v4f;

#define SEQ   2048
#define NDIM  256
#define NH    24     // 3*H heads total
#define CDIM  32     // qk channel dim per head
#define LOG2E 1.44269504088896340736f

// ---------------------------------------------------------------------------
// Workspace layout (requires ~78 MB):
//   K16      [24][2048][32]  f16   3.0 MB
//   Q16      [24][2048][32]  f16   3.0 MB
//   Vt       [24][256][2048] f16  24.0 MB   (value matrix, d-major for B-operand loads)
//   partial  [24][256][2048] f32  48.0 MB   (per-head outputs, summed by reduce)
// ---------------------------------------------------------------------------
#define K16_OFF   0
#define Q16_OFF   (24u*2048u*32u*2u)
#define VT_OFF    (2u*24u*2048u*32u*2u)
#define PART_OFF  (VT_OFF + 24u*256u*2048u*2u)

// ---------------------------------------------------------------------------
// Kernel 1: K/Q projections (small GEMMs: 256/64/4 inner dims) -> f16,
// stored head-major so the attention kernel's operand loads are coalesced.
// grid (2048, 6), block 256.  blockIdx.y: 0-1 nodes, 2-3 aux, 4-5 rot.
// ---------------------------------------------------------------------------
__global__ void __launch_bounds__(256) kq_proj_kernel(
    const float* __restrict__ nodes, const float* __restrict__ aux,
    const float* __restrict__ rot,
    const float* __restrict__ Wn, const float* __restrict__ bn,
    const float* __restrict__ Wa, const float* __restrict__ ba,
    const float* __restrict__ Wr,
    _Float16* __restrict__ K16, _Float16* __restrict__ Q16) {
  __shared__ float row[256];
  const int s     = blockIdx.x;
  const int b     = blockIdx.y;      // 0..5
  const int group = b >> 1;          // 0 nodes, 1 aux, 2 rot
  const int tid   = threadIdx.x;
  const int in_dim = (group == 0) ? 256 : (group == 1 ? 64 : 4);
  const float* src = (group == 0) ? nodes + (size_t)s * 256
                   : (group == 1) ? aux   + (size_t)s * 64
                                  : rot   + (size_t)s * 4;
  if (tid < in_dim) row[tid] = src[tid];
  __syncthreads();

  const int hc      = b * 256 + tid;        // 0..1535 output column
  const int h_local = (hc & 511) >> 6;      // 0..7
  const int col     = hc & 63;              // 0..63 (k: 0..31, q: 32..63)
  const int wrow    = h_local * 64 + col;   // row of this group's W
  const float* W = (group == 0) ? Wn : (group == 1) ? Wa : Wr;
  const float* wr = W + (size_t)wrow * in_dim;
  float acc = 0.f;
  for (int k = 0; k < in_dim; ++k) acc = fmaf(row[k], wr[k], acc);
  if (group == 0)      acc += bn[wrow];
  else if (group == 1) acc += ba[wrow];

  const int h = group * 8 + h_local;
  const int c = col & 31;
  _Float16* dst = (col >= 32) ? Q16 : K16;
  dst[((size_t)h * SEQ + s) * CDIM + c] = (_Float16)acc;
}

// ---------------------------------------------------------------------------
// Kernel 2: value projection via WMMA:  val = nodes @ W_val^T + b_val,
// written TRANSPOSED as Vt[h][d][s] (f16) so the attention PV B-operand
// is a contiguous 32B load per lane.  Two output groups are processed per
// iteration with independent accumulators / B-register sets so the f16 WMMA's
// 4-coexec WAR hazard window is filled with useful work instead of v_nops.
// grid (16, 24), block 256 (8 waves). Wave w handles rows s0 = bx*128+w*16.
// ---------------------------------------------------------------------------
__global__ void __launch_bounds__(256) val_proj_kernel(
    const float* __restrict__ nodes, const float* __restrict__ Wv,
    const float* __restrict__ bv, _Float16* __restrict__ Vt) {
  const int lane = threadIdx.x & 31;
  const int wave = threadIdx.x >> 5;
  const int s0   = blockIdx.x * 128 + wave * 16;
  const int h    = blockIdx.y;           // head = output slab of 256 cols
  const int n    = lane & 15;
  const bool hiL = (lane >= 16);

  // A operand (nodes rows s0..s0+15), 8 K-chunks of 32, f32 -> f16.
  // ISA A layout: lane<16 halves e<8 -> K=e, e>=8 -> K=16+(e-8);
  //               lane>=16: K=8+e / K=24+(e-8).  (+ kc*32)
  v16h a[8];
  const float* arow = nodes + (size_t)(s0 + n) * 256;
  #pragma unroll
  for (int kc = 0; kc < 8; ++kc) {
    const int kb1 = kc * 32 + (hiL ? 8 : 0);
    const int kb2 = kc * 32 + (hiL ? 24 : 16);
    #pragma unroll
    for (int e = 0; e < 8; ++e) {
      a[kc][e]     = (_Float16)arow[kb1 + e];
      a[kc][e + 8] = (_Float16)arow[kb2 + e];
    }
  }

  #pragma unroll 1
  for (int og = 0; og < 16; og += 2) {
    const int o0 = h * 256 + og * 16;        // group A columns
    const int o1 = o0 + 16;                  // group B columns
    const float* wrow0 = Wv + (size_t)(o0 + n) * 256 + (hiL ? 8 : 0);
    const float* wrow1 = Wv + (size_t)(o1 + n) * 256 + (hiL ? 8 : 0);
    const float bias0  = bv[o0 + n];
    const float bias1  = bv[o1 + n];
    v8f acc0, acc1;
    #pragma unroll
    for (int r = 0; r < 8; ++r) { acc0[r] = bias0; acc1[r] = bias1; }
    #pragma unroll
    for (int kc = 0; kc < 8; ++kc) {
      v16h bm0, bm1;
      #pragma unroll
      for (int e = 0; e < 16; ++e) {
        bm0[e] = (_Float16)wrow0[kc * 32 + e + (e >= 8 ? 8 : 0)];
        bm1[e] = (_Float16)wrow1[kc * 32 + e + (e >= 8 ? 8 : 0)];
      }
      acc0 = __builtin_amdgcn_wmma_f32_16x16x32_f16(false, a[kc], false, bm0,
                                                    (short)0, acc0, false, false);
      acc1 = __builtin_amdgcn_wmma_f32_16x16x32_f16(false, a[kc], false, bm1,
                                                    (short)0, acc1, false, false);
    }
    // C/D layout: acc[r] is (s = s0 + r + 8*hiL, o = o? + n) -> contiguous in s.
    v8h out0, out1;
    #pragma unroll
    for (int r = 0; r < 8; ++r) { out0[r] = (_Float16)acc0[r]; out1[r] = (_Float16)acc1[r]; }
    _Float16* dst0 = Vt + ((size_t)h * 256 + og * 16 + n) * SEQ + s0 + (hiL ? 8 : 0);
    _Float16* dst1 = dst0 + (size_t)16 * SEQ;
    *(v8h*)dst0 = out0;
    *(v8h*)dst1 = out1;
  }
}

// ---------------------------------------------------------------------------
// Kernel 3: fused flash attention per (16-row i-block, head). 1 wave / block.
// logits tile = K(16x32) @ Q^T via WMMA; exp2-based online softmax with a
// uniform rescale-skip vote; P -> LDS -> A-layout; 16 PV WMMAs per 32-j chunk
// accumulate the full [16,256] f32 output tile held in 128 VGPRs.
// grid (128, 24), block 32.
// ---------------------------------------------------------------------------
__global__ void __launch_bounds__(32) attn_kernel(
    const _Float16* __restrict__ K16, const _Float16* __restrict__ Q16,
    const _Float16* __restrict__ Vt, float* __restrict__ partial) {
  __shared__ __align__(32) _Float16 P[16 * 32];
  const int lane = threadIdx.x & 31;
  const int n    = lane & 15;
  const bool hiL = (lane >= 16);
  const int i0   = blockIdx.x * 16;
  const int h    = blockIdx.y;
  const bool squared = (h >= 16);   // rot heads: square logits pre-softmax

  // Resident A operand: K rows i0..i0+15 in contiguous k order (k = 16*hiL + e).
  const _Float16* krow = K16 + ((size_t)h * SEQ + i0 + n) * CDIM;
  v16h ka;
  {
    v8h lo = *(const v8h*)(krow + (hiL ? 16 : 0));
    v8h hp = *(const v8h*)(krow + (hiL ? 24 : 8));
    #pragma unroll
    for (int e = 0; e < 8; ++e) { ka[e] = lo[e]; ka[e + 8] = hp[e]; }
  }

  v8f acc[16];                      // [16 d-groups][8 rows] = 16x256 f32 tile
  #pragma unroll
  for (int g = 0; g < 16; ++g)
    #pragma unroll
    for (int r = 0; r < 8; ++r) acc[g][r] = 0.f;
  float m[8], l[8];                 // running row max (base-2 domain) / sums
  #pragma unroll
  for (int r = 0; r < 8; ++r) { m[r] = -1e30f; l[r] = 0.f; }
  v8f z;
  #pragma unroll
  for (int r = 0; r < 8; ++r) z[r] = 0.f;

  const _Float16* qbase = Q16 + (size_t)h * SEQ * CDIM + (hiL ? 16 : 0);
  const _Float16* vbase = Vt  + (size_t)h * 256 * SEQ  + (hiL ? 16 : 0);

  #pragma unroll 1
  for (int j0 = 0; j0 < SEQ; j0 += 32) {
    // B operands: Q rows j0..j0+15 and j0+16..j0+31 (contiguous 32B / lane)
    v16h qb0 = *(const v16h*)(qbase + (size_t)(j0 + n) * CDIM);
    v16h qb1 = *(const v16h*)(qbase + (size_t)(j0 + 16 + n) * CDIM);
    if (j0 + 32 < SEQ) {  // gfx1250 global_prefetch of next chunk's Q and Vt
      __builtin_prefetch(qbase + (size_t)(j0 + 32 + n) * CDIM, 0, 1);
      __builtin_prefetch(vbase + (size_t)n * SEQ + j0 + 32, 0, 1);
    }

    v8f t0 = __builtin_amdgcn_wmma_f32_16x16x32_f16(false, ka, false, qb0,
                                                    (short)0, z, false, false);
    v8f t1 = __builtin_amdgcn_wmma_f32_16x16x32_f16(false, ka, false, qb1,
                                                    (short)0, z, false, false);
    if (squared) { t0 *= t0; t1 *= t1; }
    t0 *= LOG2E;                    // base-2 domain: exp(x) == exp2(x*log2e)
    t1 *= LOG2E;

    // Row max. C/D layout: t[r] is (row i0+r+8*hiL, col j0+n / j0+16+n),
    // so the row reduction is across the 16-lane half-wave.
    float cm[8];
    bool new_max = false;
    #pragma unroll
    for (int r = 0; r < 8; ++r) {
      float c = fmaxf(t0[r], t1[r]);
      c = fmaxf(c, __shfl_xor(c, 1, 16));
      c = fmaxf(c, __shfl_xor(c, 2, 16));
      c = fmaxf(c, __shfl_xor(c, 4, 16));
      c = fmaxf(c, __shfl_xor(c, 8, 16));
      cm[r] = c;
      new_max |= (c > m[r]);
    }
    // Uniform vote: only pay alpha-exp + 128-mul accumulator rescale when some
    // row's running max actually moved (rare once maxima stabilize).
    if (__any((int)new_max)) {
      v8f av;
      #pragma unroll
      for (int r = 0; r < 8; ++r) {
        const float mn = fmaxf(m[r], cm[r]);
        const float alpha = exp2f(m[r] - mn);
        m[r]  = mn;
        av[r] = alpha;
        l[r] *= alpha;
      }
      #pragma unroll
      for (int g = 0; g < 16; ++g) acc[g] *= av;
    }
    v8f p0, p1;
    #pragma unroll
    for (int r = 0; r < 8; ++r) {
      p0[r] = exp2f(t0[r] - m[r]);
      p1[r] = exp2f(t1[r] - m[r]);
      l[r] += p0[r] + p1[r];        // per-lane partial row sum
    }

    // P (C/D layout) -> LDS [16][32] f16 -> reload in A layout.
    const int prow = hiL ? 8 : 0;
    #pragma unroll
    for (int r = 0; r < 8; ++r) {
      P[(r + prow) * 32 + n]      = (_Float16)p0[r];
      P[(r + prow) * 32 + n + 16] = (_Float16)p1[r];
    }
    __syncthreads();
    v16h pa;
    {
      const _Float16* pr = P + n * 32 + (hiL ? 16 : 0);
      v8h lo = *(const v8h*)(pr);
      v8h hp = *(const v8h*)(pr + 8);
      #pragma unroll
      for (int e = 0; e < 8; ++e) { pa[e] = lo[e]; pa[e + 8] = hp[e]; }
    }
    __syncthreads();   // WAR: LDS free for next chunk's stores

    // PV: B[k=j][n=d] = Vt[h][d][j0+k] -> contiguous 32B per lane; the 16
    // WMMAs are independent accumulator chains and pipeline back-to-back.
    const _Float16* vj = vbase + j0;
    #pragma unroll
    for (int g = 0; g < 16; ++g) {
      v16h vb = *(const v16h*)(vj + (size_t)(g * 16 + n) * SEQ);
      acc[g] = __builtin_amdgcn_wmma_f32_16x16x32_f16(false, pa, false, vb,
                                                      (short)0, acc[g], false, false);
    }
  }

  // Final row sums and normalization.
  v8f inv;
  #pragma unroll
  for (int r = 0; r < 8; ++r) {
    float ls = l[r];
    ls += __shfl_xor(ls, 1, 16);
    ls += __shfl_xor(ls, 2, 16);
    ls += __shfl_xor(ls, 4, 16);
    ls += __shfl_xor(ls, 8, 16);
    inv[r] = 1.0f / ls;
  }
  // Store partial[h][d][i]; C/D rows are contiguous in i -> two v4f stores.
  #pragma unroll
  for (int g = 0; g < 16; ++g) {
    v8f o = acc[g] * inv;
    v4f lo4, hi4;
    #pragma unroll
    for (int r = 0; r < 4; ++r) { lo4[r] = o[r]; hi4[r] = o[r + 4]; }
    float* dst = partial + ((size_t)h * 256 + g * 16 + n) * SEQ + i0 + (hiL ? 8 : 0);
    *(v4f*)dst       = lo4;
    *(v4f*)(dst + 4) = hi4;
  }
}

// ---------------------------------------------------------------------------
// Kernel 4: out[i][d] = sum_h partial[h][d][i].  grid (8, 256), block 256
// (consecutive threads -> consecutive i -> coalesced loads).
// ---------------------------------------------------------------------------
__global__ void __launch_bounds__(256) reduce_kernel(
    const float* __restrict__ partial, float* __restrict__ out) {
  const int i = blockIdx.x * 256 + threadIdx.x;
  const int d = blockIdx.y;
  float s = 0.f;
  #pragma unroll
  for (int h = 0; h < NH; ++h)
    s += partial[((size_t)h * 256 + d) * SEQ + i];
  out[(size_t)i * 256 + d] = s;
}

// ---------------------------------------------------------------------------
extern "C" void kernel_launch(void* const* d_in, const int* in_sizes, int n_in,
                              void* d_out, int out_size, void* d_ws, size_t ws_size,
                              hipStream_t stream) {
  const float* nodes = (const float*)d_in[0];
  const float* aux   = (const float*)d_in[1];
  const float* rot   = (const float*)d_in[2];
  const float* Wn    = (const float*)d_in[3];
  const float* bn    = (const float*)d_in[4];
  const float* Wa    = (const float*)d_in[5];
  const float* ba    = (const float*)d_in[6];
  const float* Wr    = (const float*)d_in[7];
  const float* Wv    = (const float*)d_in[8];
  const float* bv    = (const float*)d_in[9];
  float* out = (float*)d_out;

  char* ws = (char*)d_ws;
  _Float16* K16     = (_Float16*)(ws + K16_OFF);
  _Float16* Q16     = (_Float16*)(ws + Q16_OFF);
  _Float16* Vt      = (_Float16*)(ws + VT_OFF);
  float*    partial = (float*)(ws + PART_OFF);

  kq_proj_kernel<<<dim3(SEQ, 6), dim3(256), 0, stream>>>(
      nodes, aux, rot, Wn, bn, Wa, ba, Wr, K16, Q16);
  val_proj_kernel<<<dim3(16, 24), dim3(256), 0, stream>>>(nodes, Wv, bv, Vt);
  attn_kernel<<<dim3(SEQ / 16, NH), dim3(32), 0, stream>>>(K16, Q16, Vt, partial);
  reduce_kernel<<<dim3(8, 256), dim3(256), 0, stream>>>(partial, out);
}
